// MoE_35184372088964
// MI455X (gfx1250) — compile-verified, matrix operand order checked by reading, not run on hardware
//
#include <hip/hip_runtime.h>
#include <hip/hip_bf16.h>

// ---------------- problem constants (match reference) ----------------
#define BB   4
#define SS   2048
#define DD   1024
#define EE   8
#define HH   4096
#define KTOP 2
#define TT   (BB * SS)        // 8192 tokens
#define SLOTS (TT * KTOP)     // 16384 (expert,token) assignments, exact

typedef __attribute__((ext_vector_type(16))) __bf16 v16bf;
typedef __attribute__((ext_vector_type(8)))  __bf16 v8bf;
typedef __attribute__((ext_vector_type(8)))  float  v8f;

// f32x8 -> bf16x8 (RNE) and 16B LDS store: lowers to 4x v_cvt_pk_bf16_f32 + ds_store_b128
__device__ __forceinline__ void cvt_store8(unsigned short* dst, float4 a, float4 b) {
    v8f f = { a.x, a.y, a.z, a.w, b.x, b.y, b.z, b.w };
    *(v8bf*)dst = __builtin_convertvector(f, v8bf);
}
__device__ __forceinline__ void cvt_store8s(unsigned short* dst, const float* p, int stride) {
    v8f f = { p[0 * stride], p[1 * stride], p[2 * stride], p[3 * stride],
              p[4 * stride], p[5 * stride], p[6 * stride], p[7 * stride] };
    *(v8bf*)dst = __builtin_convertvector(f, v8bf);
}
// load one 16x(K=32) fragment row-slice: 16 contiguous bf16 (32B), 16B aligned.
// Lane mapping (single point of change if HW layout differs): lane&15 picks
// row (A) / col (B), lane>>4 picks K half (0..15 vs 16..31), contiguous in LDS.
__device__ __forceinline__ v16bf ld_frag(const unsigned short* p) {
    union { uint4 q[2]; v16bf v; } u;
    const uint4* q = (const uint4*)p;
    u.q[0] = q[0]; u.q[1] = q[1];
    return u.v;
}

// ---------------- kernel 1: noisy top-2 gate + routing ----------------
__global__ void moe_gate(const float* __restrict__ x,
                         const float* __restrict__ gw,
                         const float* __restrict__ nwt,
                         const float* __restrict__ noise,
                         int* __restrict__ counts,
                         int* __restrict__ tok_idx,
                         float* __restrict__ tok_gate) {
    int t    = blockIdx.x * 8 + (threadIdx.x >> 5);   // one wave32 per token
    int lane = threadIdx.x & 31;
    if (t >= TT) return;

    float acc[EE];
#pragma unroll
    for (int e = 0; e < EE; ++e) acc[e] = 0.f;

    const float* xr = x + (size_t)t * DD;
#pragma unroll 4
    for (int j = 0; j < DD / 32; ++j) {
        int d = j * 32 + lane;                 // lanes coalesced along D
        float xv = xr[d];
        const float4 g0 = *(const float4*)(gw + (size_t)d * EE);
        const float4 g1 = *(const float4*)(gw + (size_t)d * EE + 4);
        acc[0] += xv * g0.x; acc[1] += xv * g0.y;
        acc[2] += xv * g0.z; acc[3] += xv * g0.w;
        acc[4] += xv * g1.x; acc[5] += xv * g1.y;
        acc[6] += xv * g1.z; acc[7] += xv * g1.w;
    }
#pragma unroll
    for (int e = 0; e < EE; ++e) {
        float v = acc[e];
#pragma unroll
        for (int off = 16; off > 0; off >>= 1) v += __shfl_xor(v, off, 32);
        acc[e] = v;
    }
    if (lane == 0) {
        float lg[EE];
#pragma unroll
        for (int e = 0; e < EE; ++e)
            lg[e] = acc[e] + noise[(size_t)t * EE + e] * nwt[e];
        int i0 = 0; float v0 = lg[0];
#pragma unroll
        for (int e = 1; e < EE; ++e) if (lg[e] > v0) { v0 = lg[e]; i0 = e; }
        int i1 = -1; float v1 = -3.4e38f;
#pragma unroll
        for (int e = 0; e < EE; ++e)
            if (e != i0 && lg[e] > v1) { v1 = lg[e]; i1 = e; }
        float p0 = 1.f / (1.f + __expf(v1 - v0));   // softmax over top-2
        float p1 = 1.f - p0;
        int s0 = atomicAdd(&counts[i0], 1);
        tok_idx[i0 * TT + s0] = t;  tok_gate[i0 * TT + s0] = p0;
        int s1 = atomicAdd(&counts[i1], 1);
        tok_idx[i1 * TT + s1] = t;  tok_gate[i1 * TT + s1] = p1;
    }
}

// ---------------- kernel 2: grouped GEMM  h = (Xg@w1+b1)*silu(Xg@w2+b2) ----
// block = 256 thr (8 waves). C-tile 64 rows x 64 h-cols. K-loop over D.
#define LSTR 40   // LDS row stride in ushorts (32 data + 8 pad) => 80B, 16B aligned
__global__ void moe_ffn1(const float* __restrict__ x,
                         const float* __restrict__ w1, const float* __restrict__ b1,
                         const float* __restrict__ w2, const float* __restrict__ b2,
                         const int* __restrict__ counts,
                         const int* __restrict__ tok_idx,
                         float* __restrict__ hbuf, int hc0, int HC) {
    __shared__ __align__(16) unsigned short Xa[64 * LSTR];
    __shared__ __align__(16) unsigned short W1t[64 * LSTR];
    __shared__ __align__(16) unsigned short W2t[64 * LSTR];
    __shared__ int toks[64];

    const int e  = blockIdx.y >> 7;        // / (TT/64)
    const int mt = blockIdx.y & 127;
    const int cnt = counts[e];
    const int m0 = mt * 64;
    if (m0 >= cnt) return;
    int off = 0;
    for (int i = 0; i < e; ++i) off += counts[i];  // expert base row in hbuf

    const int tid = threadIdx.x;
    const int h0 = hc0 + blockIdx.x * 64;          // global h column base
    if (tid < 64) {
        int r = m0 + tid;
        toks[tid] = tok_idx[e * TT + min(r, cnt - 1)];
    }
    __syncthreads();

    // -------- staging assignments (hoisted out of the K loop) --------
    // X tile: [m][k], 4 threads per row, float4 x2 -> packed b128 store
    const int sx_row = tid >> 2, sx_c = (tid & 3) * 8;
    const float* xsrc = x + (size_t)toks[sx_row] * DD + sx_c;
    unsigned short* xdst = &Xa[sx_row * LSTR + sx_c];
    // W tiles: LDS [n][k]; thread owns col n = tid&63, k-run (tid>>6)*8.
    // Global reads are K-strided but lane-contiguous in n => fully coalesced.
    const int sw_n = tid & 63, sw_k = tid >> 6 << 3;
    const float* w1p = w1 + (size_t)e * DD * HH + (size_t)sw_k * HH + h0 + sw_n;
    const float* w2p = w2 + (size_t)e * DD * HH + (size_t)sw_k * HH + h0 + sw_n;
    unsigned short* w1d = &W1t[sw_n * LSTR + sw_k];
    unsigned short* w2d = &W2t[sw_n * LSTR + sw_k];

    const int wave = tid >> 5, lane = tid & 31;
    const int mw = wave >> 1, nw = wave & 1;       // 4 M-tiles x 2 wave-cols
    const int half = lane >> 4, ln = lane & 15;

    v8f a1a = {0,0,0,0,0,0,0,0}, a1b = {0,0,0,0,0,0,0,0};
    v8f a2a = {0,0,0,0,0,0,0,0}, a2b = {0,0,0,0,0,0,0,0};

    for (int k0 = 0; k0 < DD; k0 += 32) {
        { // stage gathered X rows -> bf16 LDS [m][k]
            float4 f0 = *(const float4*)(xsrc + k0);
            float4 f1 = *(const float4*)(xsrc + k0 + 4);
            cvt_store8(xdst, f0, f1);
        }
        { // stage w1/w2 slabs -> LDS [n][k] (transpose folded into global reads)
            const float* p1 = w1p + (size_t)k0 * HH;
            const float* p2 = w2p + (size_t)k0 * HH;
            if (k0 + 32 < DD) __builtin_prefetch(p1 + (size_t)32 * HH, 0, 1);
            cvt_store8s(w1d, p1, HH);
            cvt_store8s(w2d, p2, HH);
        }
        __syncthreads();
        v16bf a  = ld_frag(&Xa [(mw * 16 + ln) * LSTR + half * 16]);
        v16bf b0 = ld_frag(&W1t[(nw * 32 +      ln) * LSTR + half * 16]);
        v16bf b1f= ld_frag(&W1t[(nw * 32 + 16 + ln) * LSTR + half * 16]);
        v16bf c0f= ld_frag(&W2t[(nw * 32 +      ln) * LSTR + half * 16]);
        v16bf c1f= ld_frag(&W2t[(nw * 32 + 16 + ln) * LSTR + half * 16]);
        a1a = __builtin_amdgcn_wmma_f32_16x16x32_bf16(false, a, false, b0,  (short)0, a1a, false, false);
        a1b = __builtin_amdgcn_wmma_f32_16x16x32_bf16(false, a, false, b1f, (short)0, a1b, false, false);
        a2a = __builtin_amdgcn_wmma_f32_16x16x32_bf16(false, a, false, c0f, (short)0, a2a, false, false);
        a2b = __builtin_amdgcn_wmma_f32_16x16x32_bf16(false, a, false, c1f, (short)0, a2b, false, false);
        __syncthreads();
    }
    // epilogue: SwiGLU, write h chunk (C layout: VGPR v -> M = v + 8*half, N = ln)
#pragma unroll
    for (int t2 = 0; t2 < 2; ++t2) {
        int nc = blockIdx.x * 64 + nw * 32 + t2 * 16 + ln;   // col within chunk
        float bb1 = b1[(size_t)e * HH + hc0 + nc];
        float bb2 = b2[(size_t)e * HH + hc0 + nc];
        v8f A1 = t2 ? a1b : a1a;
        v8f A2 = t2 ? a2b : a2a;
#pragma unroll
        for (int v = 0; v < 8; ++v) {
            int row = m0 + mw * 16 + half * 8 + v;           // local row in expert
            if (row < cnt) {
                float g = A1[v] + bb1;
                float s = A2[v] + bb2;
                float hv = g * (s / (1.f + __expf(-s)));     // g * silu(s)
                hbuf[(size_t)(off + row) * HC + nc] = hv;
            }
        }
    }
}

// ---------------- kernel 3: out += gate * (h @ wp + bp), atomic combine ----
__global__ void moe_ffn2(const float* __restrict__ hbuf,
                         const float* __restrict__ wp, const float* __restrict__ bp,
                         const int* __restrict__ counts,
                         const int* __restrict__ tok_idx,
                         const float* __restrict__ tok_gate,
                         float* __restrict__ out, int hc0, int HC, int addBias) {
    __shared__ __align__(16) unsigned short Ha[64 * LSTR];
    __shared__ __align__(16) unsigned short Wt[64 * LSTR];
    __shared__ int   toks[64];
    __shared__ float gts[64];

    const int e  = blockIdx.y >> 7;
    const int mt = blockIdx.y & 127;
    const int cnt = counts[e];
    const int m0 = mt * 64;
    if (m0 >= cnt) return;
    int off = 0;
    for (int i = 0; i < e; ++i) off += counts[i];

    const int tid = threadIdx.x;
    const int d0 = blockIdx.x * 64;
    if (tid < 64) {
        int r = m0 + tid;
        bool ok = r < cnt;
        toks[tid] = tok_idx[e * TT + min(r, cnt - 1)];
        gts[tid]  = ok ? tok_gate[e * TT + r] : 0.f;
    }
    __syncthreads();

    // hoisted staging bases
    const int sh_row = tid >> 2, sh_c = (tid & 3) * 8;
    const int rc = min(m0 + sh_row, cnt - 1);
    const float* hsrc = hbuf + (size_t)(off + rc) * HC + sh_c;
    unsigned short* hdst = &Ha[sh_row * LSTR + sh_c];
    const int sw_n = tid & 63, sw_k = tid >> 6 << 3;
    const float* wpp = wp + (size_t)e * HH * DD + (size_t)(hc0 + sw_k) * DD + d0 + sw_n;
    unsigned short* wdd = &Wt[sw_n * LSTR + sw_k];

    const int wave = tid >> 5, lane = tid & 31;
    const int mw = wave >> 1, nw = wave & 1;
    const int half = lane >> 4, ln = lane & 15;

    v8f acca = {0,0,0,0,0,0,0,0}, accb = {0,0,0,0,0,0,0,0};

    for (int k0 = 0; k0 < HC; k0 += 32) {
        { // stage h rows (already compacted per expert)
            float4 f0 = *(const float4*)(hsrc + k0);
            float4 f1 = *(const float4*)(hsrc + k0 + 4);
            cvt_store8(hdst, f0, f1);
        }
        { // stage wp slab -> LDS [d][k]
            const float* p1 = wpp + (size_t)k0 * DD;
            if (k0 + 32 < HC) __builtin_prefetch(p1 + (size_t)32 * DD, 0, 1);
            cvt_store8s(wdd, p1, DD);
        }
        __syncthreads();
        v16bf a  = ld_frag(&Ha[(mw * 16 + ln) * LSTR + half * 16]);
        v16bf b0 = ld_frag(&Wt[(nw * 32 +      ln) * LSTR + half * 16]);
        v16bf b1f= ld_frag(&Wt[(nw * 32 + 16 + ln) * LSTR + half * 16]);
        acca = __builtin_amdgcn_wmma_f32_16x16x32_bf16(false, a, false, b0,  (short)0, acca, false, false);
        accb = __builtin_amdgcn_wmma_f32_16x16x32_bf16(false, a, false, b1f, (short)0, accb, false, false);
        __syncthreads();
    }
#pragma unroll
    for (int t2 = 0; t2 < 2; ++t2) {
        int dcol = d0 + nw * 32 + t2 * 16 + ln;
        float bpv = addBias ? bp[(size_t)e * DD + dcol] : 0.f;
        v8f A = t2 ? accb : acca;
#pragma unroll
        for (int v = 0; v < 8; ++v) {
            int rl = mw * 16 + half * 8 + v;        // row within 64-tile
            int row = m0 + rl;
            if (row < cnt) {
                float val = gts[rl] * (A[v] + bpv);
                unsafeAtomicAdd(&out[(size_t)toks[rl] * DD + dcol], val);
            }
        }
    }
}

// ---------------- host launcher ----------------
extern "C" void kernel_launch(void* const* d_in, const int* in_sizes, int n_in,
                              void* d_out, int out_size, void* d_ws, size_t ws_size,
                              hipStream_t stream) {
    const float* x     = (const float*)d_in[0];
    const float* gw    = (const float*)d_in[1];
    const float* nwt   = (const float*)d_in[2];
    const float* noise = (const float*)d_in[3];
    const float* w1    = (const float*)d_in[4];
    const float* b1    = (const float*)d_in[5];
    const float* w2    = (const float*)d_in[6];
    const float* b2    = (const float*)d_in[7];
    const float* wp    = (const float*)d_in[8];
    const float* bp    = (const float*)d_in[9];
    float* out = (float*)d_out;

    // workspace layout: counts | tok_idx[E*T] | tok_gate[E*T] | h chunk buffer
    char* ws = (char*)d_ws;
    int*   counts   = (int*)ws;
    int*   tok_idx  = (int*)(ws + 256);
    float* tok_gate = (float*)(ws + 256 + (size_t)EE * TT * 4);
    size_t hoff = (256 + (size_t)EE * TT * 8 + 255) & ~(size_t)255;
    float* hbuf = (float*)(ws + hoff);
    size_t havail = (ws_size > hoff) ? ws_size - hoff : 0;

    int HC = 64;                                  // H handled in chunks that fit ws
    for (int c = HH; c >= 64; c >>= 1)
        if ((size_t)SLOTS * c * 4 <= havail) { HC = c; break; }

    hipMemsetAsync(out, 0, (size_t)out_size * sizeof(float), stream);
    hipMemsetAsync(counts, 0, EE * sizeof(int), stream);

    moe_gate<<<TT / 8, 256, 0, stream>>>(x, gw, nwt, noise, counts, tok_idx, tok_gate);

    for (int hc0 = 0; hc0 < HH; hc0 += HC) {
        moe_ffn1<<<dim3(HC / 64, EE * (TT / 64)), 256, 0, stream>>>(
            x, w1, b1, w2, b2, counts, tok_idx, hbuf, hc0, HC);
        moe_ffn2<<<dim3(DD / 64, EE * (TT / 64)), 256, 0, stream>>>(
            hbuf, wp, bp, counts, tok_idx, tok_gate, out, hc0, HC, hc0 == 0);
    }
}
